// HNL_90185723281715
// MI455X (gfx1250) — matrix operand order repaired and checked by reading, not compiled, not run on hardware
//
#include <hip/hip_runtime.h>

#define B_ROWS    4096
#define IN_FEATS  1024
#define OUT_FEATS 1024
#define NUM_HEADS 16
#define NUM_MEMS  4096
#define HEAD_DIM  64

typedef __attribute__((ext_vector_type(16))) __bf16 bf16x16;
typedef __attribute__((ext_vector_type(8)))  __bf16 bf16x8;
typedef __attribute__((ext_vector_type(8)))  float  f32x8;

static __device__ __forceinline__ bf16x16 join8(bf16x8 lo, bf16x8 hi) {
    bf16x16 v;
#pragma unroll
    for (int i = 0; i < 8; ++i) { v[i] = lo[i]; v[i + 8] = hi[i]; }
    return v;
}

// ---------------- f32 -> bf16 conversion ----------------
__global__ void cvt_bf16_kernel(const float* __restrict__ src, __bf16* __restrict__ dst, int n) {
    int i = blockIdx.x * blockDim.x + threadIdx.x;
    if (i < n) dst[i] = (__bf16)src[i];
}

// ------- L2-normalize rows of 64 f32 -> 64 bf16 (one wave32 per row) -------
__global__ void rownorm64_kernel(const float* __restrict__ src, __bf16* __restrict__ dst, int nrows) {
    int wave = (int)((blockIdx.x * blockDim.x + threadIdx.x) >> 5);
    int lane = threadIdx.x & 31;
    if (wave >= nrows) return;
    const float2 v = *(const float2*)(src + (size_t)wave * 64 + lane * 2);
    float ss = v.x * v.x + v.y * v.y;
    ss += __shfl_xor(ss, 1, 32);
    ss += __shfl_xor(ss, 2, 32);
    ss += __shfl_xor(ss, 4, 32);
    ss += __shfl_xor(ss, 8, 32);
    ss += __shfl_xor(ss, 16, 32);
    float inv = rsqrtf(ss);
    __bf16* o = dst + (size_t)wave * 64 + lane * 2;
    o[0] = (__bf16)(v.x * inv);
    o[1] = (__bf16)(v.y * inv);
}

// ---- normalize memories, emit BOTH layouts: [h][m][64] and transposed [h][64][m] ----
__global__ void memnorm_kernel(const float* __restrict__ src,
                               __bf16* __restrict__ dst,
                               __bf16* __restrict__ dstT) {
    int wave = (int)((blockIdx.x * blockDim.x + threadIdx.x) >> 5); // h*4096 + m
    int lane = threadIdx.x & 31;
    if (wave >= NUM_HEADS * NUM_MEMS) return;
    int h = wave >> 12;
    int m = wave & (NUM_MEMS - 1);
    const float2 v = *(const float2*)(src + (size_t)wave * 64 + lane * 2);
    float ss = v.x * v.x + v.y * v.y;
    ss += __shfl_xor(ss, 1, 32);
    ss += __shfl_xor(ss, 2, 32);
    ss += __shfl_xor(ss, 4, 32);
    ss += __shfl_xor(ss, 8, 32);
    ss += __shfl_xor(ss, 16, 32);
    float inv = rsqrtf(ss);
    __bf16 b0 = (__bf16)(v.x * inv);
    __bf16 b1 = (__bf16)(v.y * inv);
    __bf16* o = dst + (size_t)wave * 64 + lane * 2;
    o[0] = b0;
    o[1] = b1;
    __bf16* oT = dstT + ((size_t)h * HEAD_DIM + lane * 2) * NUM_MEMS + m;
    oT[0] = b0;
    oT[NUM_MEMS] = b1;
}

// ---------------- q = x @ W^T : WMMA bf16, one wave per 16x16 tile ----------------
__global__ void __launch_bounds__(256) qproj_wmma_kernel(const __bf16* __restrict__ X,
                                                         const __bf16* __restrict__ W,
                                                         float* __restrict__ Q) {
    const int lane = threadIdx.x & 31;
    const int wave = threadIdx.x >> 5;
    const int gw   = blockIdx.x * 8 + wave;
    const int bt   = gw >> 6;   // 256 batch tiles
    const int ot   = gw & 63;   // 64 output-feature tiles
    const int b0   = bt * 16, o0 = ot * 16;
    const int l15  = lane & 15;
    const int hi   = lane >> 4;

    f32x8 acc = {};
    // A layout: lane l15 = row M; hi half selects K-base (0/8); elems 0..7 = K kb..kb+7, 8..15 = K kb+16..kb+23
    const __bf16* xrow = X + (size_t)(b0 + l15) * IN_FEATS + hi * 8;
    // B layout: lane l15 = col N; same K order as A (hi half K-base 0/8; elems 0..7 = kb..kb+7, 8..15 = kb+16..kb+23)
    const __bf16* wrow = W + (size_t)(o0 + l15) * IN_FEATS + hi * 8;

    for (int k = 0; k < IN_FEATS; k += 32) {
        bf16x16 a = join8(*(const bf16x8*)(xrow + k), *(const bf16x8*)(xrow + k + 16));
        bf16x16 b = join8(*(const bf16x8*)(wrow + k), *(const bf16x8*)(wrow + k + 16));
        acc = __builtin_amdgcn_wmma_f32_16x16x32_bf16(false, a, false, b, (short)0, acc, false, false);
    }
    // D layout: VGPR r -> row r (lanes 0-15) / r+8 (lanes 16-31), col = l15
    float* qout = Q + (size_t)(b0 + hi * 8) * OUT_FEATS + o0 + l15;
#pragma unroll
    for (int r = 0; r < 8; ++r) qout[(size_t)r * OUT_FEATS] = acc[r];
}

// -------- fused cosine-attention: scores + bounded-softmax + value combine --------
// Scores are cosine similarities (s <= 1), so softmax uses the FIXED max 1.0:
//   softmax(s) = exp(s-1) / sum(exp(s-1))   -- exact, overflow-safe, no running max.
// O accumulates the unnormalized combine; the denominator is a per-lane partial
// (each lane owns 4 fixed columns per tile) reduced across the half-wave once at
// the end. One wave per (head, 16-row batch tile); M-tile = 64 memories/iteration.
__global__ void __launch_bounds__(256) attn_wmma_kernel(const __bf16* __restrict__ Qn,
                                                        const __bf16* __restrict__ Mn,
                                                        const __bf16* __restrict__ MnT,
                                                        float* __restrict__ Out) {
    __shared__ __bf16 Pl[8][16][72];     // per-wave 16x64 P staging (+pad, 16B-aligned rows)

    const int lane = threadIdx.x & 31;
    const int wave = threadIdx.x >> 5;
    const int gw   = blockIdx.x * 8 + wave;   // 0..4095
    const int head = gw >> 8;                 // 16 heads
    const int bt   = gw & 255;                // 256 batch tiles
    const int b0   = bt * 16;
    const int l15  = lane & 15;
    const int hi   = lane >> 4;

    // Loop-invariant A operands: q rows b0+l15, this head's 64 features (K=64 -> two K=32 operands)
    const __bf16* qrow = Qn + (size_t)(b0 + l15) * OUT_FEATS + head * HEAD_DIM + hi * 8;
    bf16x16 A0 = join8(*(const bf16x8*)(qrow + 0),  *(const bf16x8*)(qrow + 16));
    bf16x16 A1 = join8(*(const bf16x8*)(qrow + 32), *(const bf16x8*)(qrow + 48));

    float l_i[8];                 // per-lane partial softmax denominators
    f32x8 O[4];
    f32x8 z = {};
#pragma unroll
    for (int r = 0; r < 8; ++r) l_i[r] = 0.0f;
#pragma unroll
    for (int t = 0; t < 4; ++t) O[t] = z;

    const __bf16* mrow = Mn  + (size_t)head * NUM_MEMS * HEAD_DIM;   // [m][d]
    const __bf16* mcol = MnT + (size_t)head * HEAD_DIM * NUM_MEMS;   // [d][m]

    for (int m0 = 0; m0 < NUM_MEMS; m0 += 64) {
        if (m0 + 64 < NUM_MEMS) {
            __builtin_prefetch(mrow + (size_t)(m0 + 64 + l15) * HEAD_DIM, 0, 1);
            __builtin_prefetch(mrow + (size_t)(m0 + 96 + l15) * HEAD_DIM, 0, 1);
        }

        // ---- scores: four 16x16 tiles covering memories m0..m0+63, K=64 ----
        f32x8 S[4];
#pragma unroll
        for (int u = 0; u < 4; ++u) {
            const __bf16* r = mrow + (size_t)(m0 + u * 16 + l15) * HEAD_DIM + hi * 8;
            bf16x16 bk0 = join8(*(const bf16x8*)(r + 0),  *(const bf16x8*)(r + 16));
            bf16x16 bk1 = join8(*(const bf16x8*)(r + 32), *(const bf16x8*)(r + 48));
            f32x8 s = z;
            s = __builtin_amdgcn_wmma_f32_16x16x32_bf16(false, A0, false, bk0, (short)0, s, false, false);
            s = __builtin_amdgcn_wmma_f32_16x16x32_bf16(false, A1, false, bk1, (short)0, s, false, false);
            S[u] = s;
        }

        // ---- bounded softmax numerators: p = exp(s - 1); accumulate per-lane denom ----
#pragma unroll
        for (int r = 0; r < 8; ++r) {
            float e0 = __expf(S[0][r] - 1.0f);
            float e1 = __expf(S[1][r] - 1.0f);
            float e2 = __expf(S[2][r] - 1.0f);
            float e3 = __expf(S[3][r] - 1.0f);
            int row = r + 8 * hi;
            Pl[wave][row][l15]      = (__bf16)e0;   // P in D-layout, row-major in LDS
            Pl[wave][row][16 + l15] = (__bf16)e1;
            Pl[wave][row][32 + l15] = (__bf16)e2;
            Pl[wave][row][48 + l15] = (__bf16)e3;
            l_i[r] += (e0 + e1) + (e2 + e3);
        }

        // ---- P: re-read in A-layout (wave-internal LDS RAW; HW keeps DS in order) ----
        bf16x16 Pa0 = join8(*(const bf16x8*)(&Pl[wave][l15][hi * 8]),
                            *(const bf16x8*)(&Pl[wave][l15][hi * 8 + 16]));
        bf16x16 Pa1 = join8(*(const bf16x8*)(&Pl[wave][l15][32 + hi * 8]),
                            *(const bf16x8*)(&Pl[wave][l15][32 + hi * 8 + 16]));

        // ---- O += P[16x64] . V[64x16] per feature tile; V read row-contiguous from [d][m] ----
#pragma unroll
        for (int t = 0; t < 4; ++t) {
            const __bf16* c = mcol + (size_t)(t * 16 + l15) * NUM_MEMS + m0 + hi * 8;
            bf16x16 v0 = join8(*(const bf16x8*)(c + 0),  *(const bf16x8*)(c + 16));
            bf16x16 v1 = join8(*(const bf16x8*)(c + 32), *(const bf16x8*)(c + 48));
            O[t] = __builtin_amdgcn_wmma_f32_16x16x32_bf16(false, Pa0, false, v0, (short)0, O[t], false, false);
            O[t] = __builtin_amdgcn_wmma_f32_16x16x32_bf16(false, Pa1, false, v1, (short)0, O[t], false, false);
        }
    }

    // ---- one final cross-lane reduction of the denominators, then scaled store ----
    const int orow = b0 + 8 * hi;
#pragma unroll
    for (int r = 0; r < 8; ++r) {
        float l = l_i[r];
        l += __shfl_xor(l, 1, 16);
        l += __shfl_xor(l, 2, 16);
        l += __shfl_xor(l, 4, 16);
        l += __shfl_xor(l, 8, 16);
        float scale = 8.0f / l;                       // sqrt(HEAD_DIM) / denom
#pragma unroll
        for (int t = 0; t < 4; ++t) {
            Out[(size_t)(orow + r) * OUT_FEATS + head * HEAD_DIM + t * 16 + l15] =
                O[t][r] * scale;
        }
    }
}

extern "C" void kernel_launch(void* const* d_in, const int* in_sizes, int n_in,
                              void* d_out, int out_size, void* d_ws, size_t ws_size,
                              hipStream_t stream) {
    const float* x   = (const float*)d_in[0];
    const float* W_q = (const float*)d_in[1];
    const float* mem = (const float*)d_in[2];
    float* out = (float*)d_out;

    char* ws = (char*)d_ws;
    size_t off = 0;
    __bf16* x_bf  = (__bf16*)(ws + off); off += (size_t)B_ROWS * IN_FEATS * 2;
    __bf16* w_bf  = (__bf16*)(ws + off); off += (size_t)OUT_FEATS * IN_FEATS * 2;
    float*  q_f   = (float*) (ws + off); off += (size_t)B_ROWS * OUT_FEATS * 4;
    __bf16* q_bf  = (__bf16*)(ws + off); off += (size_t)B_ROWS * OUT_FEATS * 2;
    __bf16* m_bf  = (__bf16*)(ws + off); off += (size_t)NUM_HEADS * NUM_MEMS * HEAD_DIM * 2;
    __bf16* mT_bf = (__bf16*)(ws + off); off += (size_t)NUM_HEADS * NUM_MEMS * HEAD_DIM * 2;

    int nx = B_ROWS * IN_FEATS;
    cvt_bf16_kernel<<<(nx + 255) / 256, 256, 0, stream>>>(x, x_bf, nx);
    int nw = OUT_FEATS * IN_FEATS;
    cvt_bf16_kernel<<<(nw + 255) / 256, 256, 0, stream>>>(W_q, w_bf, nw);

    memnorm_kernel<<<(NUM_HEADS * NUM_MEMS) / 8, 256, 0, stream>>>(mem, m_bf, mT_bf);

    qproj_wmma_kernel<<<2048, 256, 0, stream>>>(x_bf, w_bf, q_f);

    rownorm64_kernel<<<(B_ROWS * NUM_HEADS) / 8, 256, 0, stream>>>(q_f, q_bf, B_ROWS * NUM_HEADS);

    attn_wmma_kernel<<<512, 256, 0, stream>>>(q_bf, m_bf, mT_bf, out);
}
